// Model_57019985821939
// MI455X (gfx1250) — matrix operand order repaired and checked by reading, not compiled, not run on hardware
//
#include <hip/hip_runtime.h>

// ---------------------------------------------------------------------------
// Types / helpers
// ---------------------------------------------------------------------------
typedef __attribute__((ext_vector_type(16))) __bf16 v16bf;
typedef __attribute__((ext_vector_type(8)))  float  v8f;

#define DEV __device__ __forceinline__

constexpr int Sn = 128, Tn = 128, Bn = 64, En = 512, Hn = 512, Hhn = 256, OUTn = 512;

DEV unsigned short f2bf(float x) {
  unsigned int u = __builtin_bit_cast(unsigned int, x);
  return (unsigned short)((u + 0x7FFFu + ((u >> 16) & 1u)) >> 16);
}
DEV float bf2f(unsigned short u) {
  return __builtin_bit_cast(float, (unsigned int)u << 16);
}
DEV float sigmoidf(float x) { return 1.0f / (1.0f + __expf(-x)); }

union Frag { unsigned int u[8]; v16bf bf; };

// A fragment: 16x32 (MxK) bf16, row-major source (ld = lda elements).
// ISA 7.12.2: lane holds row m = lane%16; VGPR v holds K pair
// k0 = (v&4?16:0) + (lane/16)*8 + (v&3)*2  -> two contiguous 16B runs per lane.
DEV v16bf load_a(const unsigned short* A, int lda, int mbase, int kbase, int lane) {
  int m = mbase + (lane & 15);
  int half = lane >> 4;
  Frag f;
#pragma unroll
  for (int v = 0; v < 8; ++v) {
    int k = kbase + ((v & 4) ? 16 : 0) + half * 8 + (v & 3) * 2;
    f.u[v] = *(const unsigned int*)(A + (size_t)m * lda + k);
  }
  return f.bf;
}

// B fragment: 32x16 (KxN) from weight W[N,K] row-major (computing x@W^T).
// Lane holds col n = lane%16; VGPR v holds K pair
// k0 = (v&4?16:0) + (lane/16)*8 + (v&3)*2  -> two contiguous 16B runs per lane.
DEV v16bf load_b(const unsigned short* W, int ldw, int nbase, int kbase, int lane) {
  int n = nbase + (lane & 15);
  int half = lane >> 4;
  Frag f;
#pragma unroll
  for (int v = 0; v < 8; ++v) {
    int k = kbase + ((v & 4) ? 16 : 0) + half * 8 + (v & 3) * 2;
    f.u[v] = *(const unsigned int*)(W + (size_t)n * ldw + k);
  }
  return f.bf;
}

DEV v8f wmma_bf(v16bf a, v16bf b, v8f c) {
  return __builtin_amdgcn_wmma_f32_16x16x32_bf16(false, a, false, b, (short)0, c,
                                                 false, false);
}

// ---------------------------------------------------------------------------
// Small utility kernels
// ---------------------------------------------------------------------------
__global__ void cvt_bf16(const float* __restrict__ s, unsigned short* __restrict__ d, int n) {
  int i = blockIdx.x * 256 + threadIdx.x;
  if (i < n) d[i] = f2bf(s[i]);
}

__global__ void zero_u16(unsigned short* __restrict__ d, int n) {
  int i = blockIdx.x * 256 + threadIdx.x;
  if (i < n) d[i] = 0;
}

// out[s][b][e] = emb[tok[b*L + s]][e], converted to bf16.  L = S or T (=128).
__global__ void embed_gather(const int* __restrict__ tok, const float* __restrict__ emb,
                             unsigned short* __restrict__ out, int L) {
  int i = blockIdx.x * 256 + threadIdx.x;  // L*Bn*En = 4194304
  if (i >= L * Bn * En) return;
  int e = i & (En - 1);
  int sb = i >> 9;
  int b = sb & (Bn - 1);
  int s = sb >> 6;
  out[i] = f2bf(emb[(size_t)tok[b * L + s] * En + e]);
}

// ---------------------------------------------------------------------------
// Batched GEMM with 2 biases: C[M,N] = A[M,K] @ W[N,K]^T + b1 + b2
// One wave computes a 16(M) x 64(N) strip; block = 8 waves.
// grid.x = (M/16)*(N/64)/8 exactly (no tail).
// ---------------------------------------------------------------------------
__global__ __launch_bounds__(256) void gemm_bias(
    const unsigned short* __restrict__ A, int K, const unsigned short* __restrict__ W,
    int N, const float* __restrict__ b1, const float* __restrict__ b2,
    float* __restrict__ C) {
  int tid = threadIdx.x, lane = tid & 31, wid = tid >> 5;
  int gw = blockIdx.x * 8 + wid;
  int nb64 = N >> 6;
  int mt = gw / nb64;
  int nb = (gw % nb64) * 64;
  v8f acc[4] = {};
  for (int kb = 0; kb < K; kb += 32) {
    v16bf a = load_a(A, K, mt * 16, kb, lane);
#pragma unroll
    for (int j = 0; j < 4; ++j)
      acc[j] = wmma_bf(a, load_b(W, K, nb + j * 16, kb, lane), acc[j]);
  }
  int half = lane >> 4;
#pragma unroll
  for (int j = 0; j < 4; ++j) {
    int n = nb + j * 16 + (lane & 15);
    float bias = b1[n] + b2[n];
#pragma unroll
    for (int v = 0; v < 8; ++v) {
      int m = mt * 16 + half * 8 + v;
      C[(size_t)m * N + n] = acc[j][v] + bias;
    }
  }
}

// ---------------------------------------------------------------------------
// Encoder bidirectional LSTM recurrence, one block per direction (grid = 2).
// G[2,S,B,4*Hh] holds x@W_ih^T + b_ih + b_hh (precomputed, bias folded).
// h kept as bf16 in double-buffered LDS (WMMA A source); c lives in registers
// (static per-lane element ownership).  Wave = (mhalf, cgrp): two M-tiles
// (rows mhalf*32..+31) x 64 h-columns (cgrp*64..), all 4 gates -> B fragments
// are register-reused across the two M-tiles (halves L2 weight traffic).
// One barrier per step.
// ---------------------------------------------------------------------------
__global__ __launch_bounds__(256) void enc_lstm(
    const float* __restrict__ G,             // [2,S,B,1024]
    const unsigned short* __restrict__ Whh,  // [2,1024,256] bf16
    unsigned short* __restrict__ out,        // [S,B,512] bf16, cols d*256+..
    float* __restrict__ fin_h, float* __restrict__ fin_c,  // [64,512]
    unsigned short* __restrict__ fin_hbf) {                // [64,512]
  __shared__ unsigned short hbf[2][64 * 256];
  int tid = threadIdx.x, lane = tid & 31, wid = tid >> 5;
  int mhalf = wid & 1, cgrp = wid >> 1;
  int d = blockIdx.x;
  const float* Gd = G + (size_t)d * Sn * Bn * 1024;
  const unsigned short* Wd = Whh + (size_t)d * 1024 * 256;
  for (int i = tid; i < 64 * 256; i += 256) hbf[0][i] = 0;
  float creg[64];
#pragma unroll
  for (int i = 0; i < 64; ++i) creg[i] = 0.0f;
  __syncthreads();

  int cur = 0;
  for (int si = 0; si < Sn; ++si) {
    int s = d ? (Sn - 1 - si) : si;  // dir 1 runs reversed
    const float* Gs = Gd + (size_t)s * Bn * 1024;
#pragma unroll
    for (int ct = 0; ct < 4; ++ct) {
      int ncol = cgrp * 64 + ct * 16;
      v8f acc[2][4] = {};
      for (int kb = 0; kb < 256; kb += 32) {
        v16bf a0 = load_a(hbf[cur], 256, mhalf * 32, kb, lane);
        v16bf a1 = load_a(hbf[cur], 256, mhalf * 32 + 16, kb, lane);
#pragma unroll
        for (int g = 0; g < 4; ++g) {
          v16bf bfr = load_b(Wd, 256, g * 256 + ncol, kb, lane);
          acc[0][g] = wmma_bf(a0, bfr, acc[0][g]);
          acc[1][g] = wmma_bf(a1, bfr, acc[1][g]);
        }
      }
      int n = ncol + (lane & 15);
      int half = lane >> 4;
#pragma unroll
      for (int mi = 0; mi < 2; ++mi) {
#pragma unroll
        for (int v = 0; v < 8; ++v) {
          int m = mhalf * 32 + mi * 16 + half * 8 + v;  // batch row
          const float* grow = Gs + (size_t)m * 1024;
          float gi = acc[mi][0][v] + grow[n];
          float gf = acc[mi][1][v] + grow[256 + n];
          float gg = acc[mi][2][v] + grow[512 + n];
          float go = acc[mi][3][v] + grow[768 + n];
          int ci = (ct * 2 + mi) * 8 + v;
          float cn = sigmoidf(gf) * creg[ci] + sigmoidf(gi) * tanhf(gg);
          creg[ci] = cn;
          float hn = sigmoidf(go) * tanhf(cn);
          unsigned short hb = f2bf(hn);
          hbf[cur ^ 1][m * 256 + n] = hb;
          out[((size_t)s * Bn + m) * Hn + d * Hhn + n] = hb;
          if (si == Sn - 1) {  // final state (reverse dir ends at s=0: correct)
            size_t idx = (size_t)m * Hn + d * Hhn + n;
            fin_h[idx] = hn;
            fin_c[idx] = cn;
            fin_hbf[idx] = hb;
          }
        }
      }
    }
    __syncthreads();  // old-h reads done & new-h visible
    cur ^= 1;
  }
}

// ---------------------------------------------------------------------------
// Decoder LSTM cell: gates = [A1a|A1b]@W1^T + A2@W2^T + b1 + b2, cell update.
// W1[2048, Ka+Kb], W2[2048, K2]; gate rows ordered i|f|g|o blocks of 512.
// grid = 8 blocks; block owns 64 h-columns.  Wave = (mhalf, cgrp): two
// M-tiles x 16 columns, all 4 gates (B fragments reused across M-tiles).
// h bf16 is double-buffered ACROSS kernel launches (hbf_out never read here).
// The input concat [emb|feed] is fused via the (A1a,Ka)/(A1b,Kb) segments.
// ---------------------------------------------------------------------------
__global__ __launch_bounds__(256) void dec_cell(
    const unsigned short* __restrict__ A1a, int Ka,
    const unsigned short* __restrict__ A1b, int Kb,
    const unsigned short* __restrict__ W1,
    const unsigned short* __restrict__ A2, int K2,
    const unsigned short* __restrict__ W2,
    const float* __restrict__ b1, const float* __restrict__ b2,
    float* __restrict__ hst, float* __restrict__ cst,
    unsigned short* __restrict__ hbf_out) {
  int tid = threadIdx.x, lane = tid & 31, wid = tid >> 5;
  int mhalf = wid & 1, cgrp = wid >> 1;
  int ncol = blockIdx.x * 64 + cgrp * 16;
  int K1 = Ka + Kb;
  v8f acc[2][4] = {};
  for (int kb = 0; kb < Ka; kb += 32) {
    v16bf a0 = load_a(A1a, Ka, mhalf * 32, kb, lane);
    v16bf a1 = load_a(A1a, Ka, mhalf * 32 + 16, kb, lane);
#pragma unroll
    for (int g = 0; g < 4; ++g) {
      v16bf bfr = load_b(W1, K1, g * Hn + ncol, kb, lane);
      acc[0][g] = wmma_bf(a0, bfr, acc[0][g]);
      acc[1][g] = wmma_bf(a1, bfr, acc[1][g]);
    }
  }
  for (int kb = 0; kb < Kb; kb += 32) {
    v16bf a0 = load_a(A1b, Kb, mhalf * 32, kb, lane);
    v16bf a1 = load_a(A1b, Kb, mhalf * 32 + 16, kb, lane);
#pragma unroll
    for (int g = 0; g < 4; ++g) {
      v16bf bfr = load_b(W1, K1, g * Hn + ncol, Ka + kb, lane);
      acc[0][g] = wmma_bf(a0, bfr, acc[0][g]);
      acc[1][g] = wmma_bf(a1, bfr, acc[1][g]);
    }
  }
  for (int kb = 0; kb < K2; kb += 32) {
    v16bf a0 = load_a(A2, K2, mhalf * 32, kb, lane);
    v16bf a1 = load_a(A2, K2, mhalf * 32 + 16, kb, lane);
#pragma unroll
    for (int g = 0; g < 4; ++g) {
      v16bf bfr = load_b(W2, K2, g * Hn + ncol, kb, lane);
      acc[0][g] = wmma_bf(a0, bfr, acc[0][g]);
      acc[1][g] = wmma_bf(a1, bfr, acc[1][g]);
    }
  }
  int n = ncol + (lane & 15);
  int half = lane >> 4;
  float bi = b1[n] + b2[n];
  float bff = b1[Hn + n] + b2[Hn + n];
  float bg = b1[2 * Hn + n] + b2[2 * Hn + n];
  float bo = b1[3 * Hn + n] + b2[3 * Hn + n];
#pragma unroll
  for (int mi = 0; mi < 2; ++mi) {
#pragma unroll
    for (int v = 0; v < 8; ++v) {
      int m = mhalf * 32 + mi * 16 + half * 8 + v;  // batch row
      int idx = m * Hn + n;
      float gi = acc[mi][0][v] + bi;
      float gf = acc[mi][1][v] + bff;
      float gg = acc[mi][2][v] + bg;
      float go = acc[mi][3][v] + bo;
      float cn = sigmoidf(gf) * cst[idx] + sigmoidf(gi) * tanhf(gg);
      cst[idx] = cn;
      float hn = sigmoidf(go) * tanhf(cn);
      hst[idx] = hn;
      hbf_out[idx] = f2bf(hn);
    }
  }
}

// ---------------------------------------------------------------------------
// Attention + output projection for one decoder step.  grid = 64 (batch).
// scores = enc_out . h1 (masked, softmax over S), ctx = attn-weighted sum,
// out = tanh(attn_w @ [ctx; h1]).  VALU kernel (small: ~84 MFLOP/step).
// ---------------------------------------------------------------------------
__global__ __launch_bounds__(256) void attn_step(
    const float* __restrict__ h1,            // [64,512]
    const unsigned short* __restrict__ enc,  // [S,B,512] bf16
    const int* __restrict__ src_tokens,      // [B,S]
    const float* __restrict__ attn_w,        // [512,1024]
    unsigned short* __restrict__ feed,       // [64,512] bf16
    unsigned short* __restrict__ outs_t) {   // [64,512] bf16
  int b = blockIdx.x, tid = threadIdx.x;
  __shared__ float h1s[Hn];
  __shared__ float sc[Sn];
  __shared__ float red[256];
  __shared__ float cat[2 * Hn];
  for (int i = tid; i < Hn; i += 256) h1s[i] = h1[b * Hn + i];
  __syncthreads();
  {  // scores: 2 threads per source position (partial dot of 256)
    int s = tid & 127, kh = tid >> 7;
    float acc = 0.0f;
    const unsigned short* e = enc + ((size_t)s * Bn + b) * Hn + kh * 256;
    const float* hh = h1s + kh * 256;
    for (int k = 0; k < 256; ++k) acc += bf2f(e[k]) * hh[k];
    red[tid] = acc;
  }
  __syncthreads();
  if (tid < Sn) {
    float v = red[tid] + red[tid + 128];
    if (src_tokens[b * Sn + tid] == 0) v = -INFINITY;  // PAD mask
    sc[tid] = v;
  }
  __syncthreads();
  red[tid] = (tid < Sn) ? sc[tid] : -INFINITY;
  __syncthreads();
  for (int o = 128; o > 0; o >>= 1) {
    if (tid < o) red[tid] = fmaxf(red[tid], red[tid + o]);
    __syncthreads();
  }
  float mx = red[0];
  __syncthreads();
  float ssum = 0.0f;
  if (tid < Sn) {
    ssum = __expf(sc[tid] - mx);
    sc[tid] = ssum;
  }
  red[tid] = ssum;
  __syncthreads();
  for (int o = 128; o > 0; o >>= 1) {
    if (tid < o) red[tid] += red[tid + o];
    __syncthreads();
  }
  float inv = 1.0f / red[0];
  __syncthreads();
  for (int hc = tid; hc < Hn; hc += 256) {
    float acc = 0.0f;
    for (int s = 0; s < Sn; ++s)
      acc += sc[s] * bf2f(enc[((size_t)s * Bn + b) * Hn + hc]);
    cat[hc] = acc * inv;  // ctx
    cat[Hn + hc] = h1s[hc];
  }
  __syncthreads();
  for (int o = tid; o < Hn; o += 256) {
    float acc = 0.0f;
    const float* wr = attn_w + (size_t)o * (2 * Hn);
    for (int k = 0; k < 2 * Hn; ++k) acc += wr[k] * cat[k];
    float v = tanhf(acc);
    feed[b * Hn + o] = f2bf(v);
    outs_t[b * Hn + o] = f2bf(v);
  }
}

// ---------------------------------------------------------------------------
// Final projection: res[b][o][t] = outs[(t*64+b)] . pred_w[o] + pred_b[o].
// WMMA GEMM, M = T*B = 8192, N = 512, K = 512; scatter stores to [B,OUT,T].
// grid.x = (8192/16)*(512/64)/8 = 512.
// ---------------------------------------------------------------------------
__global__ __launch_bounds__(256) void pred_gemm(
    const unsigned short* __restrict__ A,  // [8192,512] bf16
    const unsigned short* __restrict__ W,  // [512,512] bf16
    const float* __restrict__ bias, float* __restrict__ out) {
  int tid = threadIdx.x, lane = tid & 31, wid = tid >> 5;
  int gw = blockIdx.x * 8 + wid;
  int mt = gw >> 3;
  int nb = (gw & 7) * 64;
  v8f acc[4] = {};
  for (int kb = 0; kb < Hn; kb += 32) {
    v16bf a = load_a(A, Hn, mt * 16, kb, lane);
#pragma unroll
    for (int j = 0; j < 4; ++j)
      acc[j] = wmma_bf(a, load_b(W, Hn, nb + j * 16, kb, lane), acc[j]);
  }
  int half = lane >> 4;
#pragma unroll
  for (int j = 0; j < 4; ++j) {
    int n = nb + j * 16 + (lane & 15);
    float bv = bias[n];
#pragma unroll
    for (int v = 0; v < 8; ++v) {
      int r = mt * 16 + half * 8 + v;
      int b = r & (Bn - 1), t = r >> 6;
      out[((size_t)b * OUTn + n) * Tn + t] = acc[j][v] + bv;
    }
  }
}

// ---------------------------------------------------------------------------
// Host orchestration
// ---------------------------------------------------------------------------
extern "C" void kernel_launch(void* const* d_in, const int* in_sizes, int n_in,
                              void* d_out, int out_size, void* d_ws, size_t ws_size,
                              hipStream_t stream) {
  (void)in_sizes; (void)n_in; (void)out_size; (void)ws_size;
  const int* src_tokens = (const int*)d_in[0];
  const int* dst = (const int*)d_in[2];
  const float* emb_in = (const float*)d_in[3];
  const float* emb_out = (const float*)d_in[4];
  const float* enc_w_ih0 = (const float*)d_in[5];  // [2,1024,512]
  const float* enc_w_ih1 = (const float*)d_in[6];  // [2,1024,512]
  const float* enc_w_hh = (const float*)d_in[7];   // [2,2,1024,256]
  const float* enc_b_ih = (const float*)d_in[8];   // [2,2,1024]
  const float* enc_b_hh = (const float*)d_in[9];
  const float* dec_w_ih0 = (const float*)d_in[10];  // [2048,1024]
  const float* dec_w_hh0 = (const float*)d_in[11];  // [2048,512]
  const float* dec_b_ih0 = (const float*)d_in[12];
  const float* dec_b_hh0 = (const float*)d_in[13];
  const float* dec_w_ih1 = (const float*)d_in[14];  // [2048,512]
  const float* dec_w_hh1 = (const float*)d_in[15];  // [2048,512]
  const float* dec_b_ih1 = (const float*)d_in[16];
  const float* dec_b_hh1 = (const float*)d_in[17];
  const float* attn_w = (const float*)d_in[18];  // [512,1024]
  const float* pred_w = (const float*)d_in[19];  // [512,512]
  const float* pred_b = (const float*)d_in[20];

  char* p = (char*)d_ws;
  auto alloc = [&](size_t bytes) -> char* {
    char* r = p;
    p += (bytes + 255) & ~(size_t)255;
    return r;
  };
  // bf16 weights
  unsigned short* wb_eih0 = (unsigned short*)alloc((size_t)2 * 1024 * 512 * 2);
  unsigned short* wb_eih1 = (unsigned short*)alloc((size_t)2 * 1024 * 512 * 2);
  unsigned short* wb_ehh = (unsigned short*)alloc((size_t)4 * 1024 * 256 * 2);
  unsigned short* wb_dih0 = (unsigned short*)alloc((size_t)2048 * 1024 * 2);
  unsigned short* wb_dhh0 = (unsigned short*)alloc((size_t)2048 * 512 * 2);
  unsigned short* wb_dih1 = (unsigned short*)alloc((size_t)2048 * 512 * 2);
  unsigned short* wb_dhh1 = (unsigned short*)alloc((size_t)2048 * 512 * 2);
  unsigned short* wb_pred = (unsigned short*)alloc((size_t)512 * 512 * 2);
  // activations
  unsigned short* enc_x = (unsigned short*)alloc((size_t)Sn * Bn * En * 2);
  unsigned short* dst_emb = (unsigned short*)alloc((size_t)Tn * Bn * En * 2);
  float* G = (float*)alloc((size_t)2 * Sn * Bn * 1024 * 4);  // per dir (reused per layer)
  unsigned short* l0out = (unsigned short*)alloc((size_t)Sn * Bn * Hn * 2);
  unsigned short* enc_out = (unsigned short*)alloc((size_t)Sn * Bn * Hn * 2);
  float* h0 = (float*)alloc((size_t)Bn * Hn * 4);
  float* c0 = (float*)alloc((size_t)Bn * Hn * 4);
  float* h1 = (float*)alloc((size_t)Bn * Hn * 4);
  float* c1 = (float*)alloc((size_t)Bn * Hn * 4);
  unsigned short* h0bf[2] = {(unsigned short*)alloc((size_t)Bn * Hn * 2),
                             (unsigned short*)alloc((size_t)Bn * Hn * 2)};
  unsigned short* h1bf[2] = {(unsigned short*)alloc((size_t)Bn * Hn * 2),
                             (unsigned short*)alloc((size_t)Bn * Hn * 2)};
  unsigned short* feed = (unsigned short*)alloc((size_t)Bn * Hn * 2);
  unsigned short* outs = (unsigned short*)alloc((size_t)Tn * Bn * Hn * 2);

  auto cvt = [&](const float* s, unsigned short* d, int n) {
    cvt_bf16<<<(n + 255) / 256, 256, 0, stream>>>(s, d, n);
  };
  cvt(enc_w_ih0, wb_eih0, 2 * 1024 * 512);
  cvt(enc_w_ih1, wb_eih1, 2 * 1024 * 512);
  cvt(enc_w_hh, wb_ehh, 4 * 1024 * 256);
  cvt(dec_w_ih0, wb_dih0, 2048 * 1024);
  cvt(dec_w_hh0, wb_dhh0, 2048 * 512);
  cvt(dec_w_ih1, wb_dih1, 2048 * 512);
  cvt(dec_w_hh1, wb_dhh1, 2048 * 512);
  cvt(pred_w, wb_pred, 512 * 512);

  embed_gather<<<(Sn * Bn * En + 255) / 256, 256, 0, stream>>>(src_tokens, emb_in, enc_x, Sn);
  embed_gather<<<(Tn * Bn * En + 255) / 256, 256, 0, stream>>>(dst, emb_out, dst_emb, Tn);
  zero_u16<<<(Bn * Hn + 255) / 256, 256, 0, stream>>>(feed, Bn * Hn);

  // ---- Encoder layer 0 ----
  // G[d] = enc_x @ W_ih0[d]^T + b_ih + b_hh   (M=8192, N=1024, K=512)
  int gblocks = (Sn * Bn / 16) * (1024 / 64) / 8;  // 1024
  for (int d = 0; d < 2; ++d) {
    gemm_bias<<<gblocks, 256, 0, stream>>>(
        enc_x, En, wb_eih0 + (size_t)d * 1024 * 512, 1024,
        enc_b_ih + (size_t)d * 1024, enc_b_hh + (size_t)d * 1024,
        G + (size_t)d * Sn * Bn * 1024);
  }
  enc_lstm<<<2, 256, 0, stream>>>(G, wb_ehh, l0out, h0, c0, h0bf[0]);

  // ---- Encoder layer 1 ----
  for (int d = 0; d < 2; ++d) {
    gemm_bias<<<gblocks, 256, 0, stream>>>(
        l0out, Hn, wb_eih1 + (size_t)d * 1024 * 512, 1024,
        enc_b_ih + (size_t)(2 + d) * 1024, enc_b_hh + (size_t)(2 + d) * 1024,
        G + (size_t)d * Sn * Bn * 1024);
  }
  enc_lstm<<<2, 256, 0, stream>>>(G, wb_ehh + (size_t)2 * 1024 * 256, enc_out, h1, c1,
                                  h1bf[0]);

  // ---- Decoder: 128 steps (concat fused into dec_cell via A1a/A1b) ----
  int pp = 0;
  for (int t = 0; t < Tn; ++t) {
    dec_cell<<<8, 256, 0, stream>>>(dst_emb + (size_t)t * Bn * En, En, feed, Hn,
                                    wb_dih0, h0bf[pp], Hn, wb_dhh0, dec_b_ih0,
                                    dec_b_hh0, h0, c0, h0bf[1 - pp]);
    dec_cell<<<8, 256, 0, stream>>>(h0bf[1 - pp], Hn, h0bf[1 - pp], 0, wb_dih1,
                                    h1bf[pp], Hn, wb_dhh1, dec_b_ih1, dec_b_hh1, h1,
                                    c1, h1bf[1 - pp]);
    attn_step<<<Bn, 256, 0, stream>>>(h1, enc_out, src_tokens, attn_w, feed,
                                      outs + (size_t)t * Bn * Hn);
    pp ^= 1;
  }

  // ---- Final projection ----
  pred_gemm<<<(Tn * Bn / 16) * (OUTn / 64) / 8, 256, 0, stream>>>(
      outs, wb_pred, pred_b, (float*)d_out);
}
